// Transformer_79087527788827
// MI455X (gfx1250) — compile-verified, matrix operand order checked by reading, not run on hardware
//
#include <hip/hip_runtime.h>
#include <hip/hip_bf16.h>
#include <cstddef>

// ---------------- model constants ----------------
constexpr int cV  = 128;            // vocab
constexpr int cD  = 384;            // n_embd
constexpr int cH  = 6;              // heads
constexpr int cHD = 64;             // head dim
constexpr int cL  = 6;              // layers
constexpr int cT  = 1024;           // seq
constexpr int cB  = 8;              // batch
constexpr int cFF = 1536;           // 4*D
constexpr int cM  = cB * cT;        // 8192 token rows

// ---------------- WMMA helpers ----------------
typedef __attribute__((ext_vector_type(16))) __bf16 v16bf;
typedef __attribute__((ext_vector_type(8)))  float  v8f;

#define WMMA_BF16(a, b, c) \
  __builtin_amdgcn_wmma_f32_16x16x32_bf16(false, (a), false, (b), (short)0, (c), false, false)

// 16-bit A/B operand lane->K mapping for V_WMMA_*_16X16X32_* (ISA 7.12.2):
// lanes 0-15:  elems 0..7 -> K=0..7,  elems 8..15 -> K=16..23
// lanes 16-31: elems 0..7 -> K=8..15, elems 8..15 -> K=24..31
__device__ __forceinline__ int kmap(int e, int lane) {
  return (e & 7) + ((e >> 3) << 4) + ((lane >> 4) << 3);
}

// ---------------- embedding ----------------
__global__ void embed_kernel(const int* __restrict__ idx,
                             const float* __restrict__ pe,
                             const float* __restrict__ de,
                             const float* __restrict__ pos,
                             float* __restrict__ x,
                             float* __restrict__ dur) {
  int i = blockIdx.x * blockDim.x + threadIdx.x;
  if (i >= cM * cD) return;
  int row = i / cD, c = i % cD;
  int t = row % cT;
  int ip = idx[row * 2 + 0];
  int id = idx[row * 2 + 1];
  float p = pos[t * cD + c];
  x[i]   = pe[ip * cD + c] + p;
  dur[i] = de[id * cD + c] + p;
}

// ---------------- layernorm (one block per row) ----------------
__global__ __launch_bounds__(128) void ln_kernel(const float* __restrict__ x,
                                                 const float* __restrict__ g,
                                                 const float* __restrict__ bb,
                                                 float* __restrict__ y) {
  __shared__ float red[128];
  int row = blockIdx.x, tid = threadIdx.x;
  const float* xr = x + (size_t)row * cD;
  float s = 0.f;
  for (int i = tid; i < cD; i += 128) s += xr[i];
  red[tid] = s; __syncthreads();
  for (int o = 64; o > 0; o >>= 1) { if (tid < o) red[tid] += red[tid + o]; __syncthreads(); }
  float mean = red[0] / cD;
  __syncthreads();
  float s2 = 0.f;
  for (int i = tid; i < cD; i += 128) { float d = xr[i] - mean; s2 += d * d; }
  red[tid] = s2; __syncthreads();
  for (int o = 64; o > 0; o >>= 1) { if (tid < o) red[tid] += red[tid + o]; __syncthreads(); }
  float inv = rsqrtf(red[0] / cD + 1e-5f);
  float* yr = y + (size_t)row * cD;
  for (int i = tid; i < cD; i += 128) yr[i] = (xr[i] - mean) * inv * g[i] + bb[i];
}

// ---------------- generic bf16-WMMA GEMM ----------------
// C[M,N] = A[M,K] @ B[K,N] (+bias) (+resid) (relu?)
// 256 threads = 8 waves (4x2); block tile 128x64; wave tile 32x32 -> 4 WMMAs/K-step.
// M % 128 == 0, N % 64 == 0, K % 32 == 0 (holds for all GEMMs here).
__global__ __launch_bounds__(256) void gemm_kernel(const float* __restrict__ A,
                                                   const float* __restrict__ Bw,
                                                   float* __restrict__ C,
                                                   const float* __restrict__ bias,
                                                   const float* __restrict__ resid,
                                                   int Ndim, int Kdim, int relu) {
  __shared__ __bf16 As[128][40];  // [m][k], row stride 80B keeps 8B alignment for packed stores
  __shared__ __bf16 Bs[64][40];   // [n][k] (transposed)
  int tid  = threadIdx.x;
  int lane = tid & 31, wv = tid >> 5;
  int wm = wv & 3, wn = wv >> 2;          // 4x2 wave grid -> 128x64
  int m0 = blockIdx.y * 128, n0 = blockIdx.x * 64;
  v8f acc00 = {}, acc01 = {}, acc10 = {}, acc11 = {};

  for (int kt = 0; kt < Kdim; kt += 32) {
    // A tile 128x32: float4 loads, packed 4xbf16 (8B) LDS stores
    for (int i = tid; i < 128 * 8; i += 256) {
      int r = i >> 3, c = (i & 7) << 2;
      const float4 a4 = *(const float4*)&A[(size_t)(m0 + r) * Kdim + kt + c];
      __bf16 b4[4] = { (__bf16)a4.x, (__bf16)a4.y, (__bf16)a4.z, (__bf16)a4.w };
      *(uint2*)&As[r][c] = *(const uint2*)b4;
    }
    // B tile 32x64: float4 loads along N, transposed scalar LDS stores
    for (int i = tid; i < 32 * 16; i += 256) {
      int k = i >> 4, n = (i & 15) << 2;
      const float4 b4 = *(const float4*)&Bw[(size_t)(kt + k) * Ndim + n0 + n];
      Bs[n + 0][k] = (__bf16)b4.x;
      Bs[n + 1][k] = (__bf16)b4.y;
      Bs[n + 2][k] = (__bf16)b4.z;
      Bs[n + 3][k] = (__bf16)b4.w;
    }
    __syncthreads();
    v16bf af0, af1, bf0, bf1;
    int row0 = wm * 32 + (lane & 15);
    int col0 = wn * 32 + (lane & 15);
#pragma unroll
    for (int e = 0; e < 16; ++e) {
      int kk = kmap(e, lane);
      af0[e] = As[row0][kk];
      af1[e] = As[row0 + 16][kk];
      bf0[e] = Bs[col0][kk];
      bf1[e] = Bs[col0 + 16][kk];
    }
    acc00 = WMMA_BF16(af0, bf0, acc00);
    acc01 = WMMA_BF16(af0, bf1, acc01);
    acc10 = WMMA_BF16(af1, bf0, acc10);
    acc11 = WMMA_BF16(af1, bf1, acc11);
    __syncthreads();
  }

  // C/D layout: lanes 0-15 -> N=lane, M rows 0..7; lanes 16-31 -> N=lane-16, M rows 8..15
  int rbase = m0 + wm * 32 + ((lane >> 4) << 3);
  int cl = lane & 15;
  int gn0 = n0 + wn * 32 + cl;
  int gn1 = gn0 + 16;
  v8f accs[2][2] = {{acc00, acc01}, {acc10, acc11}};
#pragma unroll
  for (int rm = 0; rm < 2; ++rm) {
#pragma unroll
    for (int r = 0; r < 8; ++r) {
      int gm = rbase + rm * 16 + r;
      float v0 = accs[rm][0][r], v1 = accs[rm][1][r];
      if (bias)  { v0 += bias[gn0]; v1 += bias[gn1]; }
      if (resid) { v0 += resid[(size_t)gm * Ndim + gn0]; v1 += resid[(size_t)gm * Ndim + gn1]; }
      if (relu)  { v0 = fmaxf(v0, 0.f); v1 = fmaxf(v1, 0.f); }
      C[(size_t)gm * Ndim + gn0] = v0;
      C[(size_t)gm * Ndim + gn1] = v1;
    }
  }
}

// ---------------- causal attention ----------------
// grid: (T/16, B*H). One block owns 16 query rows of one (b,h).
// Full score row (<=1024 keys) kept in LDS; Q.K^T and P.V via bf16 WMMA.
__global__ __launch_bounds__(256) void attn_kernel(const float* __restrict__ q,
                                                   const float* __restrict__ k,
                                                   const float* __restrict__ v,
                                                   float* __restrict__ att) {
  __shared__ float sc[16][cT];          // 64 KB scores
  __shared__ float red[16][16];
  __shared__ float rowsum[16];
  __shared__ float rowmax[16];
  int tid = threadIdx.x, lane = tid & 31, wv = tid >> 5;
  int qt = blockIdx.x, bh = blockIdx.y;
  int b = bh / cH, hh = bh % cH, hoff = hh * cHD;
  int qbase = qt * 16;
  size_t rowg = (size_t)b * cT;
  const float scale = rsqrtf((float)cD);    // reference scales by n_embd^-0.5

  // Q fragments (16 rows x HD=64 -> two K=32 chunks), same for every wave.
  v16bf qa0, qa1;
  {
    const float* qr = q + (rowg + qbase + (lane & 15)) * cD + hoff;
#pragma unroll
    for (int e = 0; e < 16; ++e) {
      int kk = kmap(e, lane);
      qa0[e] = (__bf16)qr[kk];
      qa1[e] = (__bf16)qr[kk + 32];
    }
  }

  int nk32 = qbase / 32 + 1;   // 32-key chunks covering causal extent
  int nk16 = nk32 * 2;

  // S = Q.K^T, tiles of 16 keys distributed over 8 waves
  for (int j = wv; j < nk16; j += 8) {
    int kb = j * 16;
    v16bf kb0, kb1;
    const float* kr = k + (rowg + kb + (lane & 15)) * cD + hoff;
#pragma unroll
    for (int e = 0; e < 16; ++e) {
      int kk = kmap(e, lane);
      kb0[e] = (__bf16)kr[kk];
      kb1[e] = (__bf16)kr[kk + 32];
    }
    v8f s = {};
    s = WMMA_BF16(qa0, kb0, s);
    s = WMMA_BF16(qa1, kb1, s);
    int cl = lane & 15;
    int rb = (lane >> 4) << 3;
#pragma unroll
    for (int r = 0; r < 8; ++r) {
      int qi = rb + r;
      int ky = kb + cl;
      sc[qi][ky] = (ky <= qbase + qi) ? s[r] * scale : -3.0e38f;
    }
  }
  __syncthreads();

  // masked softmax over kcount keys; 16 threads per row
  int kcount = nk16 * 16;
  int rr = tid >> 4, jj = tid & 15;
  float m = -3.0e38f;
  for (int ky = jj; ky < kcount; ky += 16) m = fmaxf(m, sc[rr][ky]);
  red[rr][jj] = m; __syncthreads();
  if (jj == 0) {
    float mm = red[rr][0];
    for (int i2 = 1; i2 < 16; ++i2) mm = fmaxf(mm, red[rr][i2]);
    rowmax[rr] = mm;
  }
  __syncthreads();
  float mx = rowmax[rr];
  float sum = 0.f;
  for (int ky = jj; ky < kcount; ky += 16) {
    float e = __expf(sc[rr][ky] - mx);
    sc[rr][ky] = e;
    sum += e;
  }
  red[rr][jj] = sum; __syncthreads();
  if (jj == 0) {
    float ss = 0.f;
    for (int i2 = 0; i2 < 16; ++i2) ss += red[rr][i2];
    rowsum[rr] = ss;
  }
  __syncthreads();

  // att = P @ V : waves 0..3 each own one 16x16 output tile (HD = 4*16)
  if (wv < 4) {
    v8f o = {};
    int nd = wv * 16 + (lane & 15);
    for (int c = 0; c < nk32; ++c) {
      v16bf pa, vb;
#pragma unroll
      for (int e = 0; e < 16; ++e) {
        int kk = kmap(e, lane);
        pa[e] = (__bf16)sc[lane & 15][c * 32 + kk];
        vb[e] = (__bf16)v[(rowg + c * 32 + kk) * cD + hoff + nd];
      }
      o = WMMA_BF16(pa, vb, o);
    }
    int cl = lane & 15;
    int rb = (lane >> 4) << 3;
#pragma unroll
    for (int r = 0; r < 8; ++r) {
      int qi = rb + r;
      att[(rowg + qbase + qi) * cD + hoff + wv * 16 + cl] = o[r] / rowsum[qi];
    }
  }
}

// ---------------- interleave pitch-embedding / dur streams ----------------
__global__ void interleave_kernel(const float* __restrict__ le,
                                  const float* __restrict__ dur,
                                  float* __restrict__ xd) {
  int i = blockIdx.x * blockDim.x + threadIdx.x;
  if (i >= cM * cD) return;
  int row = i / cD, c = i % cD;
  xd[(size_t)row * (2 * cD) + 2 * c]     = le[i];
  xd[(size_t)row * (2 * cD) + 2 * c + 1] = dur[i];
}

// ---------------- cross-entropy (one block of V=128 threads per row) ----------------
__global__ __launch_bounds__(128) void ce_kernel(const float* __restrict__ logits,
                                                 const int* __restrict__ targets,
                                                 int comp, float* __restrict__ acc) {
  __shared__ float red[128];
  int row = blockIdx.x, tid = threadIdx.x;
  const float* lr = logits + (size_t)row * cV;
  float xv = lr[tid];
  red[tid] = xv; __syncthreads();
  for (int o = 64; o > 0; o >>= 1) { if (tid < o) red[tid] = fmaxf(red[tid], red[tid + o]); __syncthreads(); }
  float mx = red[0]; __syncthreads();
  red[tid] = __expf(xv - mx); __syncthreads();
  for (int o = 64; o > 0; o >>= 1) { if (tid < o) red[tid] += red[tid + o]; __syncthreads(); }
  if (tid == 0) {
    int tgt = targets[row * 2 + comp];
    float lp = lr[tgt] - mx - __logf(red[0]);
    atomicAdd(acc, -lp);
  }
}

__global__ void zero_acc_kernel(float* acc) { acc[0] = 0.f; acc[1] = 0.f; }

__global__ void finalize_kernel(const float* __restrict__ acc, float* __restrict__ out) {
  float lp = acc[0] / (float)cM;
  float ld = acc[1] / (float)cM;
  out[0] = lp + ld;   // loss
  out[1] = lp;        // loss_pair[0]
  out[2] = ld;        // loss_pair[1]
}

// ---------------- host orchestration ----------------
static inline void launch_gemm(const float* A, const float* Bw, float* C,
                               const float* bias, const float* resid,
                               int N, int K, int relu, hipStream_t s) {
  dim3 grid(N / 64, cM / 128);
  gemm_kernel<<<grid, 256, 0, s>>>(A, Bw, C, bias, resid, N, K, relu);
}

extern "C" void kernel_launch(void* const* d_in, const int* in_sizes, int n_in,
                              void* d_out, int out_size, void* d_ws, size_t ws_size,
                              hipStream_t stream) {
  (void)in_sizes; (void)n_in; (void)out_size; (void)ws_size;

  const int*   idx     = (const int*)  d_in[0];
  const int*   targets = (const int*)  d_in[1];
  const float* pe      = (const float*)d_in[2];   // pitch_emb [V,D]
  const float* de      = (const float*)d_in[3];   // dur_emb   [V,D]
  const float* pos     = (const float*)d_in[4];   // pos_emb   [T,D]
  const float* wq      = (const float*)d_in[5];   // [L,D,D]
  const float* wk      = (const float*)d_in[6];
  const float* wv_     = (const float*)d_in[7];
  const float* wo      = (const float*)d_in[8];
  const float* bo      = (const float*)d_in[9];   // [L,D]
  const float* w1      = (const float*)d_in[10];  // [L,D,FF]
  const float* b1      = (const float*)d_in[11];  // [L,FF]
  const float* w2      = (const float*)d_in[12];  // [L,FF,D]
  const float* b2      = (const float*)d_in[13];  // [L,D]
  const float* g_ln1   = (const float*)d_in[14];
  const float* b_ln1   = (const float*)d_in[15];
  const float* g_ln2   = (const float*)d_in[16];
  const float* b_ln2   = (const float*)d_in[17];
  const float* g_lnf   = (const float*)d_in[18];
  const float* b_lnf   = (const float*)d_in[19];
  const float* w_hp    = (const float*)d_in[20];  // [D,V]
  const float* b_hp    = (const float*)d_in[21];  // [V]
  const float* w_l2e   = (const float*)d_in[22];  // [V,D]
  const float* b_l2e   = (const float*)d_in[23];  // [D]
  const float* w_hd    = (const float*)d_in[24];  // [2D,V]
  const float* b_hd    = (const float*)d_in[25];  // [V]

  float* ws = (float*)d_ws;
  const size_t MD = (size_t)cM * cD;
  float* xa   = ws;                                  // current residual stream
  float* dur  = ws + 1 * MD;
  float* h    = ws + 2 * MD;                         // LN output / logits_emb
  float* qb   = ws + 3 * MD;
  float* kb   = ws + 4 * MD;
  float* vb   = ws + 5 * MD;
  float* attb = ws + 6 * MD;
  float* xb   = ws + 7 * MD;                         // post-attention residual
  float* fb   = ws + 8 * MD;                         // [M,FF]
  float* xd   = fb + (size_t)cM * cFF;               // [M,2D]
  float* acc  = xd + (size_t)cM * 2 * cD;            // 2 loss accumulators

  float* out   = (float*)d_out;
  float* lp    = out;                     // logits_pitch [M,V]
  float* ld    = out + (size_t)cM * cV;   // logits_dur   [M,V]
  float* tailo = out + 2 * (size_t)cM * cV;

  // embeddings
  {
    int n = cM * cD;
    embed_kernel<<<(n + 255) / 256, 256, 0, stream>>>(idx, pe, de, pos, xa, dur);
  }

  // transformer blocks
  for (int l = 0; l < cL; ++l) {
    const float* wq_l = wq + (size_t)l * cD * cD;
    const float* wk_l = wk + (size_t)l * cD * cD;
    const float* wv_l = wv_ + (size_t)l * cD * cD;
    const float* wo_l = wo + (size_t)l * cD * cD;
    const float* bo_l = bo + (size_t)l * cD;
    const float* w1_l = w1 + (size_t)l * cD * cFF;
    const float* b1_l = b1 + (size_t)l * cFF;
    const float* w2_l = w2 + (size_t)l * cFF * cD;
    const float* b2_l = b2 + (size_t)l * cD;

    ln_kernel<<<cM, 128, 0, stream>>>(xa, g_ln1 + (size_t)l * cD, b_ln1 + (size_t)l * cD, h);
    launch_gemm(h, wq_l, qb, nullptr, nullptr, cD, cD, 0, stream);
    launch_gemm(h, wk_l, kb, nullptr, nullptr, cD, cD, 0, stream);
    launch_gemm(h, wv_l, vb, nullptr, nullptr, cD, cD, 0, stream);

    attn_kernel<<<dim3(cT / 16, cB * cH), 256, 0, stream>>>(qb, kb, vb, attb);

    launch_gemm(attb, wo_l, xb, bo_l, xa, cD, cD, 0, stream);   // x = x + att@wo + bo

    ln_kernel<<<cM, 128, 0, stream>>>(xb, g_ln2 + (size_t)l * cD, b_ln2 + (size_t)l * cD, h);
    launch_gemm(h, w1_l, fb, b1_l, nullptr, cFF, cD, 1, stream);   // relu(h@w1+b1)
    launch_gemm(fb, w2_l, xa, b2_l, xb, cD, cFF, 0, stream);       // x = x + f@w2 + b2
  }

  // heads
  ln_kernel<<<cM, 128, 0, stream>>>(xa, g_lnf, b_lnf, h);
  launch_gemm(h, w_hp, lp, b_hp, nullptr, cV, cD, 0, stream);      // logits_pitch
  launch_gemm(lp, w_l2e, h, b_l2e, nullptr, cD, cV, 0, stream);    // logits_emb (reuse h)
  {
    int n = cM * cD;
    interleave_kernel<<<(n + 255) / 256, 256, 0, stream>>>(h, dur, xd);
  }
  launch_gemm(xd, w_hd, ld, b_hd, nullptr, cV, 2 * cD, 0, stream); // logits_dur

  // losses
  zero_acc_kernel<<<1, 1, 0, stream>>>(acc);
  ce_kernel<<<cM, 128, 0, stream>>>(lp, targets, 0, acc + 0);
  ce_kernel<<<cM, 128, 0, stream>>>(ld, targets, 1, acc + 1);
  finalize_kernel<<<1, 1, 0, stream>>>(acc, tailo);
}
